// ModulatedConv2d_3968549781798
// MI455X (gfx1250) — matrix operand order repaired and checked by reading, compiled, not run on hardware
//
#include <hip/hip_runtime.h>
#include <hip/hip_bf16.h>

typedef __attribute__((ext_vector_type(16))) _Float16 v16h;
typedef __attribute__((ext_vector_type(8)))  _Float16 v8h;
typedef __attribute__((ext_vector_type(2)))  _Float16 v2h;
typedef __attribute__((ext_vector_type(8)))  float    v8f;

#define IC 512
#define OC 512
#define HW 64
#define NB 8
#define KK 4608                       // IC * 9
#define MOD_SCALE  0.04419417382415922f   // 1/sqrt(512)
#define CONV_SCALE 0.014731391274719742f  // 1/sqrt(512*9)
#define ASTR 40                       // padded LDS stride (halfs): conflict-free b128 reads
#define BSTR 40

struct F3 { float a, b, c; };         // 12B contiguous weight taps -> global_load_b96

// ---------------- Kernel 1: s'[b,ic] = CONV_SCALE * (style @ (W*MOD_SCALE)^T + bias) ----------------
__global__ void style_mod_kernel(const float* __restrict__ style,
                                 const float* __restrict__ modw,
                                 const float* __restrict__ modb,
                                 float* __restrict__ s2) {
  int t  = blockIdx.x * blockDim.x + threadIdx.x;   // 0..4095
  int b  = t >> 9;
  int ic = t & 511;
  const float* st = style + b * 512;
  const float* wr = modw + (size_t)ic * 512;
  float acc = 0.f;
  for (int j = 0; j < 512; ++j) acc += st[j] * wr[j];
  s2[t] = CONV_SCALE * (acc * MOD_SCALE + modb[ic]);
}

// ---------------- Kernel 2: demod[b,oc] = rsqrt( sum_{ic,k,k} (weight * s')^2 + eps ) ----------------
__global__ void demod_kernel(const float* __restrict__ weight,
                             const float* __restrict__ s2,
                             float* __restrict__ demod) {
  int lane = threadIdx.x & 31;
  int row  = blockIdx.x * 8 + (threadIdx.x >> 5);   // 0..4095 == b*512+oc
  int b  = row >> 9;
  int oc = row & 511;
  const float* wr = weight + (size_t)oc * KK;
  const float* sr = s2 + b * 512;
  float acc = 0.f;
  for (int e = lane; e < KK; e += 32) {
    float w = wr[e] * sr[e / 9];
    acc += w * w;
  }
  for (int off = 16; off > 0; off >>= 1) acc += __shfl_xor(acc, off, 32);
  if (lane == 0) demod[row] = rsqrtf(acc + 1e-8f);
}

// ---------------- Kernel 3: implicit-GEMM modulated conv, WMMA f32<=f16 ----------------
// grid = (h-quads=16, ocTile=4, b=8); block = 256 (8 wave32)
// Output tile: 128 oc x 256 pixels (4 image rows). K staged as (kh, ic-chunk of 32);
// B tile loaded once per stage with zero-halo pixel padding and reused for kw=0,1,2.
// Waves arranged 2(M) x 4(N): 4x4 fragments per wave -> 48 wmma per stage per wave.
__global__ __launch_bounds__(256) void modconv_wmma_kernel(
    const float* __restrict__ x, const float* __restrict__ weight,
    const float* __restrict__ s2, const float* __restrict__ demod,
    float* __restrict__ out) {
  __shared__ _Float16 As[3][128][ASTR];  // [kw][oc][ic]
  __shared__ _Float16 Bs[264][BSTR];     // [pixel-slot][ic]; slot = pr*66 + (imagecol+1)
  __shared__ float sS[512];
  __shared__ float dS[128];

  const int tid    = threadIdx.x;
  const int h0     = blockIdx.x * 4;
  const int ocBase = blockIdx.y * 128;
  const int b      = blockIdx.z;

  for (int i = tid; i < 512; i += 256) sS[i] = s2[b * 512 + i];
  for (int i = tid; i < 128; i += 256) dS[i] = demod[b * 512 + ocBase + i];

  const int lane   = tid & 31;
  const int waveId = tid >> 5;
  const int wm     = (waveId & 1) * 64;   // wave M offset: 0 or 64
  const int wn     = (waveId >> 1) * 64;  // wave N offset: 0,64,128,192
  const int l15    = lane & 15;
  const int hi16   = lane >> 4;           // 0: lanes 0-15, 1: lanes 16-31

  const v8f vzero = {0.f, 0.f, 0.f, 0.f, 0.f, 0.f, 0.f, 0.f};
  v8f acc[4][4];
  #pragma unroll
  for (int mf = 0; mf < 4; ++mf)
    #pragma unroll
    for (int nf = 0; nf < 4; ++nf) acc[mf][nf] = vzero;

  const float* xb = x + (size_t)b * IC * HW * HW;

  for (int kh = 0; kh < 3; ++kh) {
    for (int ic0 = 0; ic0 < IC; ic0 += 32) {
      __syncthreads();
      // A tiles: 3 kw taps for an ic-pair: two b96 loads, pk-converts, 3x ds_store_b32
      #pragma unroll
      for (int r = 0; r < 8; ++r) {
        int idx = r * 256 + tid;               // 0..2047
        int oc = idx >> 4, icp = (idx & 15) * 2;
        const float* wp0 = weight + (size_t)(ocBase + oc) * KK + (ic0 + icp) * 9 + kh * 3;
        F3 w0 = *(const F3*)wp0;
        F3 w1 = *(const F3*)(wp0 + 9);
        float sa = sS[ic0 + icp], sb = sS[ic0 + icp + 1];
        v2h t0; t0[0] = (_Float16)(w0.a * sa); t0[1] = (_Float16)(w1.a * sb);
        v2h t1; t1[0] = (_Float16)(w0.b * sa); t1[1] = (_Float16)(w1.b * sb);
        v2h t2; t2[0] = (_Float16)(w0.c * sa); t2[1] = (_Float16)(w1.c * sb);
        *(v2h*)&As[0][oc][icp] = t0;
        *(v2h*)&As[1][oc][icp] = t1;
        *(v2h*)&As[2][oc][icp] = t2;
      }
      // B tile: 4 image rows x 64 cols x 32 ic, ic-pairs packed, shared across kw
      #pragma unroll
      for (int r = 0; r < 16; ++r) {
        int idx = r * 256 + tid;               // 0..4095 (pixel, ic-pair)
        int pc = idx & 63;
        int pr = (idx >> 6) & 3;
        int icp = (idx >> 8) * 2;
        int hh = h0 + pr + kh - 1;
        float x0 = 0.f, x1 = 0.f;
        if ((unsigned)hh < 64u) {
          const float* xp = xb + ((size_t)(ic0 + icp) * HW + hh) * HW + pc;
          x0 = xp[0];
          x1 = xp[HW * HW];
        }
        v2h t; t[0] = (_Float16)x0; t[1] = (_Float16)x1;
        *(v2h*)&Bs[pr * 66 + pc + 1][icp] = t;
      }
      // zero halo columns (image cols -1 and 64 for all 4 rows), 128 packed stores
      if (tid < 128) {
        int icp = (tid & 15) * 2, side = (tid >> 4) & 1, pr = (tid >> 5) & 3;
        v2h z; z[0] = (_Float16)0.f; z[1] = (_Float16)0.f;
        *(v2h*)&Bs[pr * 66 + side * 65][icp] = z;
      }
      __syncthreads();

      #pragma unroll
      for (int kw = 0; kw < 3; ++kw) {
        // A fragments (ISA 16-bit A 16x32 layout: two ds_load_b128 each)
        v16h afrag[4];
        #pragma unroll
        for (int mf = 0; mf < 4; ++mf) {
          int row = wm + mf * 16 + l15;
          int kb = hi16 * 8;                   // lanes 0-15: K{0-7,16-23}; 16-31: K{8-15,24-31}
          v8h lo = *(const v8h*)&As[kw][row][kb];
          v8h hi = *(const v8h*)&As[kw][row][kb + 16];
          #pragma unroll
          for (int i = 0; i < 8; ++i) { afrag[mf][i] = lo[i]; afrag[mf][i + 8] = hi[i]; }
        }
        // B fragments one at a time (keeps live VGPRs low); kw shift = pixel-slot offset
        #pragma unroll
        for (int nf = 0; nf < 4; ++nf) {
          int p = wn + nf * 16 + l15;          // logical pixel 0..255
          int slot = (p >> 6) * 66 + (p & 63) + kw;
          int kb = hi16 * 8;                   // lanes 0-15: K{0-7,16-23}; 16-31: K{8-15,24-31}
          v8h lo = *(const v8h*)&Bs[slot][kb];
          v8h hi = *(const v8h*)&Bs[slot][kb + 16];
          v16h bfrag;
          #pragma unroll
          for (int i = 0; i < 8; ++i) { bfrag[i] = lo[i]; bfrag[i + 8] = hi[i]; }
          #pragma unroll
          for (int mf = 0; mf < 4; ++mf)
            acc[mf][nf] = __builtin_amdgcn_wmma_f32_16x16x32_f16(
                false, afrag[mf], false, bfrag, (short)0, acc[mf][nf],
                false, false);
        }
      }
    }
  }

  // Epilogue: apply per-(b,oc) demod scalar, store f32.
  // D layout: VGPR r, lane L -> row M = r + 8*hi16, col N = L&15
  #pragma unroll
  for (int mf = 0; mf < 4; ++mf) {
    #pragma unroll
    for (int nf = 0; nf < 4; ++nf) {
      int p = wn + nf * 16 + l15;
      int pr = p >> 6, pc = p & 63;
      #pragma unroll
      for (int r = 0; r < 8; ++r) {
        int ocl = wm + mf * 16 + r + hi16 * 8;
        out[(((size_t)b * OC + ocBase + ocl) * HW + h0 + pr) * HW + pc] =
            acc[mf][nf][r] * dS[ocl];
      }
    }
  }
}

extern "C" void kernel_launch(void* const* d_in, const int* in_sizes, int n_in,
                              void* d_out, int out_size, void* d_ws, size_t ws_size,
                              hipStream_t stream) {
  const float* x      = (const float*)d_in[0];   // [8,512,64,64]
  const float* style  = (const float*)d_in[1];   // [8,512]
  const float* modw   = (const float*)d_in[2];   // [512,512]
  const float* modb   = (const float*)d_in[3];   // [512]
  const float* weight = (const float*)d_in[4];   // [1,512,512,3,3]
  float* out = (float*)d_out;                    // [8,512,64,64]

  float* s2    = (float*)d_ws;                   // [8,512]  modulated scale (CONV_SCALE folded)
  float* demod = s2 + NB * IC;                   // [8,512]

  style_mod_kernel<<<16, 256, 0, stream>>>(style, modw, modb, s2);
  demod_kernel<<<512, 256, 0, stream>>>(weight, s2, demod);

  dim3 grid(HW / 4, OC / 128, NB);               // (16, 4, 8)
  modconv_wmma_kernel<<<grid, 256, 0, stream>>>(x, weight, s2, demod, out);
}
